// GeneralizedAttention_6296422056055
// MI455X (gfx1250) — matrix operand order repaired and checked by reading, not compiled, not run on hardware
//
#include <hip/hip_runtime.h>
#include <hip/hip_bf16.h>

// ---------- CDNA5 WMMA types ----------
typedef __attribute__((ext_vector_type(16))) __bf16 v16bf;
typedef __attribute__((ext_vector_type(8)))  float  v8f;
typedef __attribute__((ext_vector_type(4))) unsigned int v4u;
typedef __attribute__((ext_vector_type(8))) int v8i;
typedef __attribute__((ext_vector_type(4))) int v4i;

union Frag {
    v16bf v;
    unsigned int u[8];
};

#define WMMA_BF16(a, b, c) \
    __builtin_amdgcn_wmma_f32_16x16x32_bf16(false, (a), false, (b), (short)0, (c), false, false)

// ---------- bf16 helpers (bit-level) ----------
static __device__ __forceinline__ unsigned short f32_to_bf16(float f) {
    unsigned int u = __float_as_uint(f);
    unsigned int lsb = (u >> 16) & 1u;
    u += 0x7fffu + lsb;               // round to nearest even
    return (unsigned short)(u >> 16);
}
static __device__ __forceinline__ float bf16_to_f32(unsigned short h) {
    return __uint_as_float(((unsigned int)h) << 16);
}

// ---------- TDM (Tensor Data Mover) ----------
#if defined(__has_builtin)
#if __has_builtin(__builtin_amdgcn_tensor_load_to_lds)
#define HAVE_TDM 1
#endif
#endif
#ifndef HAVE_TDM
#define HAVE_TDM 0
#endif

#if HAVE_TDM
// 2D tile load: tile_d1 rows of tile_d0 2-byte elements, row stride = stride_elems.
// D# bit layout per CDNA5 ISA.
// This toolchain's builtin is the 6-arg form:
//   (uint32x4 g0, int32x8 g1, int32x4, int32x4, int32x8, i32 cpol)
static __device__ __forceinline__ void tdm_load_tile(unsigned lds_off, const void* gsrc,
                                                     unsigned tile_d0, unsigned tile_d1,
                                                     unsigned stride_elems)
{
    unsigned long long ga = (unsigned long long)(size_t)gsrc;
    v4u g0;
    g0[0] = 1u;                                     // count=1 (valid), user mode
    g0[1] = lds_off;                                // lds_addr (bytes)   bits 63:32
    g0[2] = (unsigned)(ga & 0xffffffffu);           // global_addr[31:0]  bits 95:64
    g0[3] = (unsigned)(((ga >> 32) & 0x1ffffffu))   // global_addr[56:32] bits 120:96
          | (2u << 30);                             // type=2 ("image")   bits 127:126
    v8i g1;
    g1[0] = (int)(1u << 16);                        // workgroup_mask=0, data_size=1 (2B)
    g1[1] = 0;                                      // abar=0, tensor_dim0[15:0]=0
    g1[2] = 0x10;                                   // tensor_dim0=1<<20 (hi16=0x10), td1 lo=0
    g1[3] = (int)(1u | (tile_d0 << 16));            // tensor_dim1=1<<16 (hi16=1), tile_dim0
    g1[4] = (int)tile_d1;                           // tile_dim1, tile_dim2=0
    g1[5] = (int)stride_elems;                      // tensor_dim0_stride[31:0]
    g1[6] = 0;
    g1[7] = 0;
    v4i z4 = {0, 0, 0, 0};
    v8i z8;
    z8[0] = 0; z8[1] = 0; z8[2] = 0; z8[3] = 0;
    z8[4] = 0; z8[5] = 0; z8[6] = 0; z8[7] = 0;
    __builtin_amdgcn_tensor_load_to_lds(g0, g1, z4, z4, z8, 0);
}
#endif

// Problem constants
#define NB      2
#define CDIM    256
#define NPIX    4096      // 64*64
#define NKV     1024      // 32*32
#define NH      8

// ============================================================================
// Kernel 1: prep — transpose/cast x -> xT bf16 [n][p][c], xkvT bf16 [n][kp][c],
//           cast weights to bf16.
// ============================================================================
__global__ void prep_kernel(const float* __restrict__ x,
                            const float* __restrict__ qw, const float* __restrict__ kw,
                            const float* __restrict__ vw, const float* __restrict__ pw,
                            unsigned short* __restrict__ xT,
                            unsigned short* __restrict__ xkvT,
                            unsigned short* __restrict__ wq, unsigned short* __restrict__ wk,
                            unsigned short* __restrict__ wv, unsigned short* __restrict__ wp)
{
    const int NX   = NB * NPIX * CDIM;       // 2,097,152
    const int NXKV = NB * NKV * CDIM;        // 524,288
    const int NW   = 4 * 65536;              // 262,144
    const int total = NX + NXKV + NW;
    for (int idx = blockIdx.x * blockDim.x + threadIdx.x; idx < total;
         idx += gridDim.x * blockDim.x) {
        if (idx < NX) {
            int c  = idx & 255;
            int p  = (idx >> 8) & 4095;
            int nb = idx >> 20;
            xT[idx] = f32_to_bf16(x[((size_t)nb * CDIM + c) * NPIX + p]);
        } else if (idx < NX + NXKV) {
            int j  = idx - NX;
            int c  = j & 255;
            int kp = (j >> 8) & 1023;
            int nb = j >> 18;
            int p  = (kp >> 5) * 128 + (kp & 31) * 2;   // stride-2 subsample
            xkvT[j] = f32_to_bf16(x[((size_t)nb * CDIM + c) * NPIX + p]);
        } else {
            int j = idx - NX - NXKV;
            int w = j >> 16;
            int e = j & 65535;
            if      (w == 0) wq[e] = f32_to_bf16(qw[e]);
            else if (w == 1) wk[e] = f32_to_bf16(kw[e]);
            else if (w == 2) wv[e] = f32_to_bf16(vw[e]);
            else             wp[e] = f32_to_bf16(pw[e]);
        }
    }
}

// ============================================================================
// Kernel 2: position features (+ geom bias folded into gpfx/gpfy tables).
// ============================================================================
__global__ void posfeat_kernel(const float* __restrict__ fcx, const float* __restrict__ fcy,
                               const float* __restrict__ gb,
                               float* __restrict__ pfx, unsigned short* __restrict__ pfyb,
                               float* __restrict__ gpfx, float* __restrict__ gpfy)
{
    __shared__ float elds[128];
    __shared__ float glds[256];
    int b    = blockIdx.x;
    int axis = b >> 11;           // 0 = x-axis, 1 = y-axis
    int rem  = b & 2047;
    int q    = rem >> 5;          // query coordinate (0..63)
    int kq   = rem & 31;          // kv coordinate (0..31)
    int t    = threadIdx.x;       // 0..255 (= head*32 + d)

    float diff = (float)q - 2.0f * (float)kq;
    if (t < 128) {
        int fr = t & 63;
        float a = diff * __powf(1000.0f, -(1.0f / 64.0f) * (float)fr);
        elds[t] = (t >> 6) ? __cosf(a) : __sinf(a);
    }
    __syncthreads();
    const float* fc = axis ? fcy : fcx;
    float acc = 0.0f;
    #pragma unroll 4
    for (int f = 0; f < 128; ++f) acc += fc[t * 128 + f] * elds[f];
    acc *= 0.70710678118654752f;  // 1/sqrt(2)

    int h = t >> 5, d = t & 31;
    if (axis == 0) pfx[(((size_t)h * 64 + q) * 32 + d) * 32 + kq] = acc;
    else           pfyb[(((size_t)h * 64 + q) * 32 + kq) * 32 + d] = f32_to_bf16(acc);

    glds[t] = gb[t] * acc;
    __syncthreads();
    if (t < 8) {
        float s = 0.0f;
        #pragma unroll
        for (int dd = 0; dd < 32; ++dd) s += glds[t * 32 + dd];
        if (axis == 0) gpfx[((size_t)t * 64 + q) * 32 + kq] = s;
        else           gpfy[((size_t)t * 64 + q) * 32 + kq] = s;
    }
}

// ============================================================================
// Kernel 3: QKV projection GEMMs via WMMA, 32x32 output tile per wave
// (2 A-frags x 2 B-frags -> 4 WMMA per k-step).
//   C[o][p] = sum_c W[o][c] * Xt[p][c]
//   MODE 0: Q  -> [n][h][p][d]   MODE 1: Kt -> [n][h][kp][d]
//   MODE 2: Vc -> [n][h][vd][kp]
// ============================================================================
template <int MODE>
__global__ void __launch_bounds__(32)
qkv_gemm_kernel(const unsigned short* __restrict__ W,
                const unsigned short* __restrict__ Xt,
                unsigned short* __restrict__ Out, int P)
{
    int tid   = threadIdx.x;
    int lane  = tid & 15;
    int khalf = tid >> 4;
    int ptiles = P >> 5;
    int b  = blockIdx.x;
    int pt = b % ptiles;
    int ot = (b / ptiles) & 7;
    int nb = b / (ptiles * 8);

    const unsigned short* A0 = W + (size_t)(ot * 32 + lane) * 256;
    const unsigned short* A1 = A0 + 16 * 256;
    const unsigned short* B0 = Xt + ((size_t)nb * P + pt * 32 + lane) * 256;
    const unsigned short* B1 = B0 + 16 * 256;

    v8f acc[2][2];
    #pragma unroll
    for (int i = 0; i < 2; ++i)
        #pragma unroll
        for (int j = 0; j < 2; ++j)
            #pragma unroll
            for (int r = 0; r < 8; ++r) acc[i][j][r] = 0.0f;

    for (int kc = 0; kc < 8; ++kc) {
        Frag a0, a1, b0, b1;
        #pragma unroll
        for (int v = 0; v < 8; ++v) {
            int ka = khalf * 16 + 2 * v;
            a0.u[v] = *(const unsigned int*)(A0 + kc * 32 + ka);
            a1.u[v] = *(const unsigned int*)(A1 + kc * 32 + ka);
            int kb = khalf * 16 + 2 * v;
            b0.u[v] = *(const unsigned int*)(B0 + kc * 32 + kb);
            b1.u[v] = *(const unsigned int*)(B1 + kc * 32 + kb);
        }
        acc[0][0] = WMMA_BF16(a0.v, b0.v, acc[0][0]);
        acc[0][1] = WMMA_BF16(a0.v, b1.v, acc[0][1]);
        acc[1][0] = WMMA_BF16(a1.v, b0.v, acc[1][0]);
        acc[1][1] = WMMA_BF16(a1.v, b1.v, acc[1][1]);
    }

    #pragma unroll
    for (int i = 0; i < 2; ++i) {
        #pragma unroll
        for (int j = 0; j < 2; ++j) {
            int p = pt * 32 + j * 16 + lane;
            #pragma unroll
            for (int r = 0; r < 8; ++r) {
                int o  = ot * 32 + i * 16 + r + khalf * 8;
                int hh = o >> 5, dd = o & 31;
                unsigned short val = f32_to_bf16(acc[i][j][r]);
                if (MODE == 0)
                    Out[(((size_t)nb * NH + hh) * NPIX + p) * 32 + dd] = val;
                else if (MODE == 1)
                    Out[(((size_t)nb * NH + hh) * NKV + p) * 32 + dd] = val;
                else
                    Out[(((size_t)nb * NH + hh) * 32 + dd) * NKV + p] = val;
            }
        }
    }
}

// ============================================================================
// Kernel 4: abk[n][h][kp] = sum_d appr_bias[h,d] * K[n,h,d,kp]
// ============================================================================
__global__ void abk_kernel(const unsigned short* __restrict__ Kt,
                           const float* __restrict__ ab, float* __restrict__ abk)
{
    int idx = blockIdx.x * 256 + threadIdx.x;
    if (idx >= NB * NH * NKV) return;
    int kp = idx & 1023;
    int h  = (idx >> 10) & 7;
    int nb = idx >> 13;
    const unsigned short* krow = Kt + (((size_t)nb * NH + h) * NKV + kp) * 32;
    float s = 0.0f;
    #pragma unroll
    for (int d = 0; d < 32; ++d) s += ab[h * 32 + d] * bf16_to_f32(krow[d]);
    abk[idx] = s;
}

// ============================================================================
// Kernel 5: fused local attention (flash-attention style, online softmax).
// One wave per (n, head, 16-query tile). K/V blocks double-buffered into LDS
// by the Tensor Data Mover; compute overlaps the next block's DMA.
// ============================================================================
__global__ void __launch_bounds__(32)
attn_kernel(const unsigned short* __restrict__ Qb,
            const unsigned short* __restrict__ Kt,
            const unsigned short* __restrict__ Vc,
            const float* __restrict__ abk,
            const float* __restrict__ pfx,
            const unsigned short* __restrict__ pfyb,
            const float* __restrict__ gpfx,
            const float* __restrict__ gpfy,
            unsigned short* __restrict__ Obuf)
{
    __shared__ float qlds[16 * 32];
    __shared__ float pxlds[16 * 32];
    __shared__ float pylds[16 * 32];
    __shared__ float abklds[1024];
    __shared__ unsigned int plds32[16 * 32 / 2];
    __shared__ unsigned short kblk[2][1024];   // 32 kpos x 32 d   (2KB each)
    __shared__ unsigned short vblk[2][1024];   // 32 vd   x 32 kpos
    unsigned short* plds = (unsigned short*)plds32;

    int tid   = threadIdx.x;
    int lane  = tid & 15;
    int khalf = tid >> 4;
    int b    = blockIdx.x;
    int tile = b & 255;
    int h    = (b >> 8) & 7;
    int nb   = b >> 11;
    int y    = tile >> 2;
    int x0   = (tile & 3) << 4;

    const unsigned short* qbase = Qb + (((size_t)nb * NH + h) * NPIX + y * 64 + x0) * 32;

    // --- Q A-fragment (registers, reused all blocks) ---
    Frag qa;
    #pragma unroll
    for (int v = 0; v < 8; ++v) {
        int ka = khalf * 16 + 2 * v;
        qa.u[v] = *(const unsigned int*)(qbase + lane * 32 + ka);
    }
    // --- Q tile to LDS (f32) for the PX VALU dot ---
    #pragma unroll 4
    for (int t = 0; t < 16; ++t) {
        int idx = t * 32 + tid;
        qlds[idx] = bf16_to_f32(qbase[idx]);
    }
    // --- abk row to LDS ---
    const float* abkb = abk + ((size_t)nb * NH + h) * NKV;
    #pragma unroll 4
    for (int t = 0; t < 32; ++t) abklds[t * 32 + tid] = abkb[t * 32 + tid];
    __syncthreads();

    // --- mask ranges (local attention, range 16, kv stride 2) ---
    int kylo = (y >= 16) ? ((y - 16) >> 1) : 0;
    int kyhi = ((y + 17) >> 1) + 1;
    if (kyhi > 32) kyhi = 32;

    const unsigned short* ktb = Kt + ((size_t)nb * NH + h) * NKV * 32;
    const unsigned short* vcb = Vc + ((size_t)nb * NH + h) * 32 * NKV;

#if HAVE_TDM
    // Preload first K/V block via Tensor Data Mover.
    tdm_load_tile((unsigned)(size_t)(void*)&kblk[0][0], ktb + (size_t)kylo * 1024, 1024, 1, 1024);
    tdm_load_tile((unsigned)(size_t)(void*)&vblk[0][0], vcb + (size_t)kylo * 32, 32, 32, 1024);
#endif

    // --- PY[i][ky] = Q . pfy[h][y]^T via WMMA (+ gpfy) ---
    const unsigned short* pyb = pfyb + ((size_t)h * 64 + y) * 32 * 32;
    const float* gpy = gpfy + ((size_t)h * 64 + y) * 32;
    #pragma unroll
    for (int half = 0; half < 2; ++half) {
        Frag bk;
        #pragma unroll
        for (int v = 0; v < 8; ++v) {
            int kb = khalf * 16 + 2 * v;
            bk.u[v] = *(const unsigned int*)(pyb + (size_t)(half * 16 + lane) * 32 + kb);
        }
        v8f c;
        #pragma unroll
        for (int r = 0; r < 8; ++r) c[r] = 0.0f;
        c = WMMA_BF16(qa.v, bk.v, c);
        float g = gpy[half * 16 + lane];
        #pragma unroll
        for (int r = 0; r < 8; ++r) {
            int i = r + khalf * 8;
            pylds[i * 32 + half * 16 + lane] = c[r] + g;
        }
    }

    // --- PX[i][kx] = sum_d q[i,d]*pfx[h,x0+i,d,kx] + gpfx (row-dependent B) ---
    {
        const float* pxb = pfx + ((size_t)h * 64 + x0) * 32 * 32;
        const float* gpx = gpfx + ((size_t)h * 64 + x0) * 32;
        for (int i = 0; i < 16; ++i) {
            float acc = gpx[i * 32 + tid];
            const float* pr = pxb + (size_t)i * 32 * 32 + tid;   // lane = kx -> coalesced
            #pragma unroll 8
            for (int d = 0; d < 32; ++d) acc += qlds[i * 32 + d] * pr[d * 32];
            pxlds[i * 32 + tid] = acc;
        }
    }
    __syncthreads();

    int kxlo[8], kxhi[8];
    #pragma unroll
    for (int r = 0; r < 8; ++r) {
        int xi = x0 + r + khalf * 8;
        kxlo[r] = (xi >= 16) ? ((xi - 16) >> 1) : 0;
        int t2 = ((xi + 17) >> 1) + 1;
        kxhi[r] = (t2 > 32) ? 32 : t2;
    }

    // --- online softmax state ---
    float mrow[8], lrow[8];
    v8f o0, o1;
    #pragma unroll
    for (int r = 0; r < 8; ++r) { mrow[r] = -3.0e38f; lrow[r] = 0.0f; o0[r] = 0.0f; o1[r] = 0.0f; }

    for (int ky = kylo; ky < kyhi; ++ky) {
        int kp0 = ky * 32;
        int pp  = (ky - kylo) & 1;
        const unsigned short* kbuf;
        const unsigned short* vbuf;

#if HAVE_TDM
        if (ky + 1 < kyhi) {
            // all ds reads of buf[1-pp] (previous block) must land before the
            // TDM engine overwrites it
            asm volatile("s_wait_dscnt 0" ::: "memory");
            tdm_load_tile((unsigned)(size_t)(void*)&kblk[1 - pp][0],
                          ktb + (size_t)(ky + 1) * 1024, 1024, 1, 1024);
            tdm_load_tile((unsigned)(size_t)(void*)&vblk[1 - pp][0],
                          vcb + (size_t)(ky + 1) * 32, 32, 32, 1024);
            __builtin_amdgcn_s_wait_tensorcnt(2);   // current block resident
        } else {
            __builtin_amdgcn_s_wait_tensorcnt(0);
        }
        asm volatile("" ::: "memory");
        kbuf = kblk[pp];
        vbuf = vblk[pp];
#else
        // Fallback: cooperative copy of current block into LDS.
        kbuf = kblk[pp];
        vbuf = vblk[pp];
        {
            unsigned int* kd = (unsigned int*)kblk[pp];
            unsigned int* vd = (unsigned int*)vblk[pp];
            const unsigned int* ks = (const unsigned int*)(ktb + (size_t)kp0 * 32);
            #pragma unroll 4
            for (int t = 0; t < 16; ++t) kd[t * 32 + tid] = ks[t * 32 + tid];
            #pragma unroll 4
            for (int t = 0; t < 16; ++t) {
                int idx = t * 32 + tid;
                int row = idx >> 4, cp = idx & 15;
                vd[idx] = *(const unsigned int*)(vcb + (size_t)row * NKV + kp0 + cp * 2);
            }
            __syncthreads();
        }
#endif

        // --- S = Q . K^T (two 16-wide kx halves), K from LDS ---
        v8f s[2];
        #pragma unroll
        for (int hx = 0; hx < 2; ++hx) {
            Frag bk;
            #pragma unroll
            for (int v = 0; v < 8; ++v) {
                int kb = khalf * 16 + 2 * v;
                bk.u[v] = *(const unsigned int*)(kbuf + (hx * 16 + lane) * 32 + kb);
            }
            v8f c;
            #pragma unroll
            for (int r = 0; r < 8; ++r) c[r] = 0.0f;
            s[hx] = WMMA_BF16(qa.v, bk.v, c);
        }

        // --- bias + mask + online softmax ---
        float e0[8], e1[8];
        #pragma unroll
        for (int r = 0; r < 8; ++r) {
            int i  = r + khalf * 8;
            float py = pylds[i * 32 + ky];
            int kx0 = lane, kx1 = 16 + lane;
            float v0 = s[0][r] + abklds[kp0 + kx0] + py + pxlds[i * 32 + kx0];
            float v1 = s[1][r] + abklds[kp0 + kx1] + py + pxlds[i * 32 + kx1];
            if (kx0 < kxlo[r] || kx0 >= kxhi[r]) v0 = -3.0e38f;
            if (kx1 < kxlo[r] || kx1 >= kxhi[r]) v1 = -3.0e38f;
            e0[r] = v0; e1[r] = v1;
        }
        #pragma unroll
        for (int r = 0; r < 8; ++r) {
            float bm = fmaxf(e0[r], e1[r]);
            #pragma unroll
            for (int off = 8; off > 0; off >>= 1) bm = fmaxf(bm, __shfl_xor(bm, off, 16));
            float nm = fmaxf(mrow[r], bm);
            float sc = __expf(mrow[r] - nm);
            float p0 = __expf(e0[r] - nm);
            float p1 = __expf(e1[r] - nm);
            float rs = p0 + p1;
            #pragma unroll
            for (int off = 8; off > 0; off >>= 1) rs += __shfl_xor(rs, off, 16);
            lrow[r] = lrow[r] * sc + rs;
            mrow[r] = nm;
            o0[r] *= sc; o1[r] *= sc;
            int i = r + khalf * 8;
            plds[i * 32 + lane]      = f32_to_bf16(p0);
            plds[i * 32 + 16 + lane] = f32_to_bf16(p1);
        }
        __syncthreads();

        // --- O += P . V, P re-fetched from LDS in A-layout, V from LDS ---
        Frag pa;
        #pragma unroll
        for (int v = 0; v < 8; ++v) {
            int ka = khalf * 16 + 2 * v;
            pa.u[v] = *(const unsigned int*)(plds + lane * 32 + ka);
        }
        #pragma unroll
        for (int vh = 0; vh < 2; ++vh) {
            Frag bv;
            #pragma unroll
            for (int v = 0; v < 8; ++v) {
                int kb = khalf * 16 + 2 * v;
                bv.u[v] = *(const unsigned int*)(vbuf + (vh * 16 + lane) * 32 + kb);
            }
            if (vh == 0) o0 = WMMA_BF16(pa.v, bv.v, o0);
            else         o1 = WMMA_BF16(pa.v, bv.v, o1);
        }
        __syncthreads();
    }

    // --- normalize and write O as [n][p][c=256] bf16 for the proj GEMM ---
    unsigned short* ob = Obuf + (((size_t)nb * NPIX) + y * 64 + x0) * 256 + h * 32;
    #pragma unroll
    for (int r = 0; r < 8; ++r) {
        int i = r + khalf * 8;
        float inv = 1.0f / lrow[r];
        ob[(size_t)i * 256 + lane]      = f32_to_bf16(o0[r] * inv);
        ob[(size_t)i * 256 + 16 + lane] = f32_to_bf16(o1[r] * inv);
    }
}

// ============================================================================
// Kernel 6: output projection + bias + gamma*out + residual, 32x32 tile/wave.
// ============================================================================
__global__ void __launch_bounds__(32)
proj_kernel(const unsigned short* __restrict__ Wp,
            const unsigned short* __restrict__ Obuf,
            const float* __restrict__ proj_b,
            const float* __restrict__ gamma,
            const float* __restrict__ x,
            float* __restrict__ out)
{
    int tid   = threadIdx.x;
    int lane  = tid & 15;
    int khalf = tid >> 4;
    int b  = blockIdx.x;
    int pt = b & 127;
    int ot = (b >> 7) & 7;
    int nb = b >> 10;

    const unsigned short* A0 = Wp + (size_t)(ot * 32 + lane) * 256;
    const unsigned short* A1 = A0 + 16 * 256;
    const unsigned short* B0 = Obuf + ((size_t)nb * NPIX + pt * 32 + lane) * 256;
    const unsigned short* B1 = B0 + 16 * 256;

    v8f acc[2][2];
    #pragma unroll
    for (int i = 0; i < 2; ++i)
        #pragma unroll
        for (int j = 0; j < 2; ++j)
            #pragma unroll
            for (int r = 0; r < 8; ++r) acc[i][j][r] = 0.0f;

    for (int kc = 0; kc < 8; ++kc) {
        Frag a0, a1, b0, b1;
        #pragma unroll
        for (int v = 0; v < 8; ++v) {
            int ka = khalf * 16 + 2 * v;
            a0.u[v] = *(const unsigned int*)(A0 + kc * 32 + ka);
            a1.u[v] = *(const unsigned int*)(A1 + kc * 32 + ka);
            int kb = khalf * 16 + 2 * v;
            b0.u[v] = *(const unsigned int*)(B0 + kc * 32 + kb);
            b1.u[v] = *(const unsigned int*)(B1 + kc * 32 + kb);
        }
        acc[0][0] = WMMA_BF16(a0.v, b0.v, acc[0][0]);
        acc[0][1] = WMMA_BF16(a0.v, b1.v, acc[0][1]);
        acc[1][0] = WMMA_BF16(a1.v, b0.v, acc[1][0]);
        acc[1][1] = WMMA_BF16(a1.v, b1.v, acc[1][1]);
    }

    float g = gamma[0];
    #pragma unroll
    for (int i = 0; i < 2; ++i) {
        #pragma unroll
        for (int j = 0; j < 2; ++j) {
            int p = pt * 32 + j * 16 + lane;
            #pragma unroll
            for (int r = 0; r < 8; ++r) {
                int o = ot * 32 + i * 16 + r + khalf * 8;
                size_t idx = ((size_t)nb * 256 + o) * NPIX + p;
                out[idx] = g * (acc[i][j][r] + proj_b[o]) + x[idx];
            }
        }
    }
}

// ============================================================================
// Host launcher
// ============================================================================
extern "C" void kernel_launch(void* const* d_in, const int* in_sizes, int n_in,
                              void* d_out, int out_size, void* d_ws, size_t ws_size,
                              hipStream_t stream)
{
    (void)in_sizes; (void)n_in; (void)out_size; (void)ws_size;

    const float* x    = (const float*)d_in[0];
    const float* qw   = (const float*)d_in[1];
    const float* kw   = (const float*)d_in[2];
    const float* vw   = (const float*)d_in[3];
    const float* fcx  = (const float*)d_in[4];
    const float* fcy  = (const float*)d_in[5];
    const float* ab   = (const float*)d_in[6];
    const float* gb   = (const float*)d_in[7];
    const float* pw   = (const float*)d_in[8];
    const float* pb   = (const float*)d_in[9];
    const float* gamma= (const float*)d_in[10];

    char* ws = (char*)d_ws;
    size_t cur = 0;
    auto alloc = [&](size_t bytes) -> char* {
        char* p = ws + cur;
        cur += (bytes + 255) & ~(size_t)255;
        return p;
    };

    unsigned short* xT    = (unsigned short*)alloc((size_t)NB * NPIX * CDIM * 2);
    unsigned short* xkvT  = (unsigned short*)alloc((size_t)NB * NKV * CDIM * 2);
    unsigned short* wqb   = (unsigned short*)alloc(65536 * 2);
    unsigned short* wkb   = (unsigned short*)alloc(65536 * 2);
    unsigned short* wvb   = (unsigned short*)alloc(65536 * 2);
    unsigned short* wpb   = (unsigned short*)alloc(65536 * 2);
    unsigned short* Qb    = (unsigned short*)alloc((size_t)NB * NH * NPIX * 32 * 2);
    unsigned short* Ktb   = (unsigned short*)alloc((size_t)NB * NH * NKV * 32 * 2);
    unsigned short* Vcb   = (unsigned short*)alloc((size_t)NB * NH * 32 * NKV * 2);
    float*          abkp  = (float*)alloc((size_t)NB * NH * NKV * 4);
    float*          pfxp  = (float*)alloc((size_t)NH * 64 * 32 * 32 * 4);
    unsigned short* pfybp = (unsigned short*)alloc((size_t)NH * 64 * 32 * 32 * 2);
    float*          gpfxp = (float*)alloc((size_t)NH * 64 * 32 * 4);
    float*          gpfyp = (float*)alloc((size_t)NH * 64 * 32 * 4);
    unsigned short* Obuf  = (unsigned short*)alloc((size_t)NB * NPIX * 256 * 2);

    prep_kernel<<<2048, 256, 0, stream>>>(x, qw, kw, vw, pw, xT, xkvT, wqb, wkb, wvb, wpb);
    posfeat_kernel<<<4096, 256, 0, stream>>>(fcx, fcy, gb, pfxp, pfybp, gpfxp, gpfyp);

    qkv_gemm_kernel<0><<<NB * 8 * (NPIX / 32), 32, 0, stream>>>(wqb, xT, Qb, NPIX);
    qkv_gemm_kernel<1><<<NB * 8 * (NKV / 32), 32, 0, stream>>>(wkb, xkvT, Ktb, NKV);
    qkv_gemm_kernel<2><<<NB * 8 * (NKV / 32), 32, 0, stream>>>(wvb, xkvT, Vcb, NKV);

    abk_kernel<<<(NB * NH * NKV + 255) / 256, 256, 0, stream>>>(Ktb, ab, abkp);

    attn_kernel<<<NB * NH * (NPIX / 16), 32, 0, stream>>>(Qb, Ktb, Vcb, abkp,
                                                          pfxp, pfybp, gpfxp, gpfyp, Obuf);

    proj_kernel<<<NB * 8 * (NPIX / 32), 32, 0, stream>>>(wpb, Obuf, pb, gamma, x,
                                                         (float*)d_out);
}
